// ReAttention_59717225283942
// MI455X (gfx1250) — matrix operand order repaired and checked by reading, not compiled, not run on hardware
//
#include <hip/hip_runtime.h>
#include <hip/hip_fp16.h>

// ---------------------------------------------------------------------------
// ReAttention (DeepViT) for MI455X / gfx1250.
// wave32 + v_wmma_f32_16x16x32_f16 + global_load_async_to_lds_b128.
// B=8, N=1024, D=512, H=8, dh=64, inner=512.
// ---------------------------------------------------------------------------

typedef __attribute__((ext_vector_type(16))) _Float16 v16h;
typedef __attribute__((ext_vector_type(8)))  _Float16 v8h;
typedef __attribute__((ext_vector_type(8)))  float    v8f;

#define NB     8
#define NS     1024
#define DMODEL 512
#define NH     8
#define DH     64
#define BNCNT  8388608.0f   /* B*N*N */

__device__ inline v8f zero_v8f() {
  v8f z;
#pragma unroll
  for (int i = 0; i < 8; ++i) z[i] = 0.0f;
  return z;
}

__device__ inline v8f wmma_f16(v16h a, v16h b, v8f c) {
  return __builtin_amdgcn_wmma_f32_16x16x32_f16(
      false, a, false, b, (short)0, c, false, false);
}

// ---- gfx1250 async copy: global -> LDS, 16 bytes per lane (ASYNCcnt) ------
__device__ inline void async_b128(unsigned int ldsoff, const void* gptr) {
  unsigned long long ga = (unsigned long long)(uintptr_t)gptr;
  asm volatile("global_load_async_to_lds_b128 %0, %1, off"
               :: "v"(ldsoff), "v"(ga) : "memory");
}
__device__ inline void wait_async0() {
  asm volatile("s_wait_asynccnt 0x0" ::: "memory");
}
__device__ inline unsigned int lds_off(const void* p) {
  // LDS aperture: flat addr low 32 bits are the LDS byte address.
  return (unsigned int)(uintptr_t)p;
}

// A fragment: 16x32 (MxK) f16 tile, row-major, stride ld (halves).
// lanes 0-15: M=lane, K {0..7,16..23}; lanes 16-31: K {8..15,24..31}.
__device__ inline v16h load_a_frag(const _Float16* As, int ld, int lane) {
  int m  = lane & 15;
  int kb = (lane & 16) ? 8 : 0;
  const _Float16* p = As + m * ld + kb;
  v8h lo = *(const v8h*)p;
  v8h hi = *(const v8h*)(p + 16);
  v16h a;
#pragma unroll
  for (int i = 0; i < 8; ++i) { a[i] = lo[i]; a[8 + i] = hi[i]; }
  return a;
}

// B fragment from fragment-major [N][K] storage, stride ld (halves).
// lanes 0-15: N=lane, K {0..7,16..23}; lanes 16-31: N=lane-16, K {8..15,24..31}.
__device__ inline v16h load_b_frag(const _Float16* BsT, int ld, int lane) {
  int n  = lane & 15;
  int kb = (lane & 16) ? 8 : 0;
  const _Float16* p = BsT + n * ld + kb;
  v8h lo = *(const v8h*)p;
  v8h hi = *(const v8h*)(p + 16);
  v16h b;
#pragma unroll
  for (int i = 0; i < 8; ++i) { b[i] = lo[i]; b[8 + i] = hi[i]; }
  return b;
}

// ---------------------------------------------------------------------------
// Kernel 0: zero BN statistic accumulators.
// ---------------------------------------------------------------------------
__global__ void k_zero_stats(float* stats) {
  if (threadIdx.x < 32) stats[threadIdx.x] = 0.0f;
}

// ---------------------------------------------------------------------------
// Kernel 1: qkv = x @ w_qkv ; q scaled by dh^-0.5 -> [B,H,N,DH] f16,
// k -> [B,H,N,DH] f16, v -> TRANSPOSED [B,H,DH,N] f16 (feeds async B tiles).
// Tile 128x128, BK=32.  grid (64,12), block 256.
// ---------------------------------------------------------------------------
__global__ __launch_bounds__(256)
void k_qkv(const float* __restrict__ x, const float* __restrict__ w,
           _Float16* __restrict__ qg, _Float16* __restrict__ kg,
           _Float16* __restrict__ vT) {
  __shared__ alignas(16) _Float16 As[128 * 32];    // [row][k]
  __shared__ alignas(16) _Float16 BsT[128 * 32];   // [col][k]  fragment-major
  const int tid  = threadIdx.x;
  const int lane = tid & 31;
  const int wid  = tid >> 5;
  const int wm   = wid >> 1;
  const int wn   = wid & 1;
  const int row0 = blockIdx.x * 128;
  const int col0 = blockIdx.y * 128;

  v8f acc[2][4];
#pragma unroll
  for (int mi = 0; mi < 2; ++mi)
#pragma unroll
    for (int ni = 0; ni < 4; ++ni) acc[mi][ni] = zero_v8f();

  for (int kk = 0; kk < DMODEL; kk += 32) {
    __syncthreads();
#pragma unroll
    for (int t = 0; t < 16; ++t) {
      int idx = tid + t * 256;
      int r = idx >> 5, c = idx & 31;
      As[idx] = (_Float16)x[(size_t)(row0 + r) * DMODEL + kk + c];
    }
#pragma unroll
    for (int t = 0; t < 16; ++t) {
      int idx = tid + t * 256;
      int c = idx >> 5, r = idx & 31;
      BsT[idx] = (_Float16)w[(size_t)(kk + r) * 1536 + col0 + c];
    }
    __syncthreads();
#pragma unroll
    for (int mi = 0; mi < 2; ++mi) {
      v16h a = load_a_frag(As + (wm * 32 + mi * 16) * 32, 32, lane);
#pragma unroll
      for (int ni = 0; ni < 4; ++ni) {
        v16h b = load_b_frag(BsT + (wn * 64 + ni * 16) * 32, 32, lane);
        acc[mi][ni] = wmma_f16(a, b, acc[mi][ni]);
      }
    }
  }

  const int rowoff = (lane & 16) ? 8 : 0;
#pragma unroll
  for (int mi = 0; mi < 2; ++mi) {
#pragma unroll
    for (int ni = 0; ni < 4; ++ni) {
      int colg = col0 + wn * 64 + ni * 16 + (lane & 15);
      int sel = colg >> 9, wi = colg & 511;
      int h = wi >> 6, d = wi & 63;
#pragma unroll
      for (int r = 0; r < 8; ++r) {
        int rowg = row0 + wm * 32 + mi * 16 + rowoff + r;
        int b_ = rowg >> 10, n_ = rowg & 1023;
        float val = acc[mi][ni][r];
        size_t bh = (size_t)(b_ * NH + h);
        if (sel == 0)      qg[(bh * NS + n_) * DH + d] = (_Float16)(val * 0.125f);
        else if (sel == 1) kg[(bh * NS + n_) * DH + d] = (_Float16)val;
        else               vT[(bh * DH + d) * NS + n_] = (_Float16)val;
      }
    }
  }
}

// ---------------------------------------------------------------------------
// Kernel 2: S[bh][n][m] = q @ k^T.  Q and K tiles staged with
// global_load_async_to_lds_b128 (both linear copies); K kept [m][d] which is
// exactly the fragment-major B layout.  grid (64,16), block 256.
// ---------------------------------------------------------------------------
__global__ __launch_bounds__(256)
void k_scores(const _Float16* __restrict__ qg, const _Float16* __restrict__ kg,
              _Float16* __restrict__ S) {
  __shared__ alignas(16) _Float16 Qs[64 * 64];     // [row][d]
  __shared__ alignas(16) _Float16 Ks[128 * 64];    // [m][d]  fragment-major
  const int tid  = threadIdx.x;
  const int lane = tid & 31;
  const int wid  = tid >> 5;
  const int rt   = wid >> 1;
  const int mh   = wid & 1;
  const int bh   = blockIdx.x;
  const int row0 = blockIdx.y * 64;
  const size_t qk_base = (size_t)bh * NS * DH;
  const size_t s_base  = (size_t)bh * NS * NS;
  const unsigned int qs0 = lds_off(Qs);
  const unsigned int ks0 = lds_off(Ks);

  // Q tile: 8KB, contiguous in qg -> 2 async b128 per thread.
#pragma unroll
  for (int t = 0; t < 2; ++t) {
    int idx = tid + t * 256;
    async_b128(qs0 + idx * 16, qg + qk_base + (size_t)row0 * DH + idx * 8);
  }

  const int rowoff = (lane & 16) ? 8 : 0;
  for (int mc = 0; mc < NS; mc += 128) {
    wait_async0();
    __syncthreads();
    // K tile: 16KB, contiguous in kg -> 4 async b128 per thread.
#pragma unroll
    for (int t = 0; t < 4; ++t) {
      int idx = tid + t * 256;
      async_b128(ks0 + idx * 16, kg + qk_base + (size_t)mc * DH + idx * 8);
    }
    wait_async0();
    __syncthreads();

    v8f acc[4];
#pragma unroll
    for (int ni = 0; ni < 4; ++ni) acc[ni] = zero_v8f();
#pragma unroll
    for (int ks = 0; ks < 2; ++ks) {
      v16h a = load_a_frag(Qs + rt * 16 * 64 + ks * 32, 64, lane);
#pragma unroll
      for (int ni = 0; ni < 4; ++ni) {
        v16h b = load_b_frag(Ks + (mh * 64 + ni * 16) * 64 + ks * 32, 64, lane);
        acc[ni] = wmma_f16(a, b, acc[ni]);
      }
    }
#pragma unroll
    for (int ni = 0; ni < 4; ++ni) {
      int colg = mc + mh * 64 + ni * 16 + (lane & 15);
#pragma unroll
      for (int r = 0; r < 8; ++r) {
        int rowg = row0 + rt * 16 + rowoff + r;
        S[s_base + (size_t)rowg * NS + colg] = (_Float16)acc[ni][r];
      }
    }
  }
}

// ---------------------------------------------------------------------------
// Kernel 3: per-row softmax stats (max, 1/sum(exp)); one wave32 per row.
// ---------------------------------------------------------------------------
__global__ __launch_bounds__(256)
void k_rowstats(const _Float16* __restrict__ S, float* __restrict__ rmax,
                float* __restrict__ rsuminv) {
  const int lane = threadIdx.x & 31;
  const int wid  = threadIdx.x >> 5;
  const size_t row = (size_t)blockIdx.x * 8 + wid;
  const _Float16* p = S + row * NS;
  float lv[32];
  float m = -3.0e38f;
#pragma unroll
  for (int j = 0; j < 32; ++j) {
    lv[j] = (float)p[lane + j * 32];
    m = fmaxf(m, lv[j]);
  }
#pragma unroll
  for (int s = 16; s > 0; s >>= 1) m = fmaxf(m, __shfl_xor(m, s, 32));
  float sum = 0.0f;
#pragma unroll
  for (int j = 0; j < 32; ++j) sum += expf(lv[j] - m);
#pragma unroll
  for (int s = 16; s > 0; s >>= 1) sum += __shfl_xor(sum, s, 32);
  if (lane == 0) { rmax[row] = m; rsuminv[row] = 1.0f / sum; }
}

// ---------------------------------------------------------------------------
// Kernel 4: softmax + 8x8 head-mix conv IN PLACE on S + BN partial sums.
// Each thread owns one (b,n,m) across all heads -> race-free in place.
// ---------------------------------------------------------------------------
__global__ __launch_bounds__(256)
void k_mix(_Float16* __restrict__ S, const float* __restrict__ rmax,
           const float* __restrict__ rsuminv, const float* __restrict__ conv_w,
           const float* __restrict__ conv_b, float* __restrict__ stats) {
  __shared__ float ssum[8], ssq[8];
  const int tid = threadIdx.x;
  if (tid < 8) { ssum[tid] = 0.0f; ssq[tid] = 0.0f; }
  __syncthreads();

  const size_t gid = (size_t)blockIdx.x * 256 + tid;
  const int b_ = (int)(gid >> 20);
  const int n_ = (int)((gid >> 10) & 1023);
  const int m_ = (int)(gid & 1023);

  float p[NH];
#pragma unroll
  for (int h = 0; h < NH; ++h) {
    size_t ridx = (size_t)(b_ * NH + h) * NS + n_;
    p[h] = expf((float)S[(ridx << 10) + m_] - rmax[ridx]) * rsuminv[ridx];
  }
  float lsum[NH], lsq[NH];
#pragma unroll
  for (int g = 0; g < NH; ++g) {
    float a = conv_b[g];
#pragma unroll
    for (int h = 0; h < NH; ++h) a += conv_w[g * NH + h] * p[h];
    S[(((size_t)(b_ * NH + g) * NS + n_) << 10) + m_] = (_Float16)a;
    lsum[g] = a;
    lsq[g]  = a * a;
  }
#pragma unroll
  for (int g = 0; g < NH; ++g) {
    atomicAdd(&ssum[g], lsum[g]);
    atomicAdd(&ssq[g], lsq[g]);
  }
  __syncthreads();
  if (tid < 8) {
    atomicAdd(&stats[tid], ssum[tid]);
    atomicAdd(&stats[8 + tid], ssq[tid]);
  }
}

// ---------------------------------------------------------------------------
// Kernel 5: BN finalize -> per-head scale/shift.
// ---------------------------------------------------------------------------
__global__ void k_bnfin(float* stats, const float* gamma, const float* beta) {
  int g = threadIdx.x;
  if (g < 8) {
    const float inv_cnt = 1.0f / BNCNT;
    float mean = stats[g] * inv_cnt;
    float var  = stats[8 + g] * inv_cnt - mean * mean;
    float inv  = rsqrtf(var + 1e-5f);
    float sc   = gamma[g] * inv;
    stats[16 + g] = sc;
    stats[24 + g] = beta[g] - mean * sc;
  }
}

// ---------------------------------------------------------------------------
// Kernel 6: out = BN(attn') @ v.  V tiles (from the [B,H,DH,N] transpose)
// are async-copied straight into fragment-major [d][m] LDS; attn tiles get
// the BN affine applied during staging.  grid (64,8), block 256.
// ---------------------------------------------------------------------------
__global__ __launch_bounds__(256)
void k_av(const _Float16* __restrict__ S, const _Float16* __restrict__ vT,
          const float* __restrict__ stats, _Float16* __restrict__ O) {
  __shared__ alignas(16) _Float16 As2[128 * 64];   // [row][m] BN-applied
  __shared__ alignas(16) _Float16 Vs[64 * 64];     // [d][m]  fragment-major
  const int tid  = threadIdx.x;
  const int lane = tid & 31;
  const int wid  = tid >> 5;
  const int bh   = blockIdx.x;
  const int row0 = blockIdx.y * 128;
  const int b_ = bh >> 3, h_ = bh & 7;
  const float sc = stats[16 + h_];
  const float sh = stats[24 + h_];
  const size_t s_base = (size_t)bh * NS * NS;
  const size_t v_base = (size_t)bh * DH * NS;      // vT rows of length NS
  const unsigned int vs0 = lds_off(Vs);

  v8f acc[4];
#pragma unroll
  for (int ni = 0; ni < 4; ++ni) acc[ni] = zero_v8f();

  for (int mc = 0; mc < NS; mc += 64) {
    __syncthreads();
    // V tile: 8KB -> 2 async b128 per thread (row d, 8-half group of m).
#pragma unroll
    for (int t = 0; t < 2; ++t) {
      int idx = tid + t * 256;
      int d = idx >> 3, m8 = idx & 7;
      async_b128(vs0 + idx * 16, vT + v_base + (size_t)d * NS + mc + m8 * 8);
    }
    // attn tile with BN affine (fp32 math) -> manual staging.
#pragma unroll
    for (int t = 0; t < 32; ++t) {
      int idx = tid + t * 256;
      int r = idx >> 6, c = idx & 63;
      float raw = (float)S[s_base + (size_t)(row0 + r) * NS + mc + c];
      As2[idx] = (_Float16)(sc * raw + sh);
    }
    wait_async0();
    __syncthreads();
#pragma unroll
    for (int ks = 0; ks < 2; ++ks) {
      v16h a = load_a_frag(As2 + wid * 16 * 64 + ks * 32, 64, lane);
#pragma unroll
      for (int ni = 0; ni < 4; ++ni) {
        v16h b = load_b_frag(Vs + ni * 16 * 64 + ks * 32, 64, lane);
        acc[ni] = wmma_f16(a, b, acc[ni]);
      }
    }
  }

  const int rowoff = (lane & 16) ? 8 : 0;
#pragma unroll
  for (int ni = 0; ni < 4; ++ni) {
    int col = h_ * DH + ni * 16 + (lane & 15);
#pragma unroll
    for (int r = 0; r < 8; ++r) {
      int n_ = row0 + wid * 16 + rowoff + r;
      O[((size_t)b_ * NS + n_) * DMODEL + col] = (_Float16)acc[ni][r];
    }
  }
}

// ---------------------------------------------------------------------------
// Kernel 7: out = O @ w_out + b_out (fp32).  O tile async-staged.
// grid (64,4), block 256.
// ---------------------------------------------------------------------------
__global__ __launch_bounds__(256)
void k_proj(const _Float16* __restrict__ O, const float* __restrict__ w,
            const float* __restrict__ bias, float* __restrict__ out) {
  __shared__ alignas(16) _Float16 As[128 * 32];    // [row][k]
  __shared__ alignas(16) _Float16 BsT[128 * 32];   // [col][k]  fragment-major
  const int tid  = threadIdx.x;
  const int lane = tid & 31;
  const int wid  = tid >> 5;
  const int wm   = wid >> 1;
  const int wn   = wid & 1;
  const int row0 = blockIdx.x * 128;
  const int col0 = blockIdx.y * 128;
  const unsigned int as0 = lds_off(As);

  v8f acc[2][4];
#pragma unroll
  for (int mi = 0; mi < 2; ++mi)
#pragma unroll
    for (int ni = 0; ni < 4; ++ni) acc[mi][ni] = zero_v8f();

  for (int kk = 0; kk < DMODEL; kk += 32) {
    __syncthreads();
    // O tile: 8KB f16 -> 2 async b128 per thread (row r, 8-half group).
#pragma unroll
    for (int t = 0; t < 2; ++t) {
      int idx = tid + t * 256;
      int r = idx >> 2, c8 = idx & 3;
      async_b128(as0 + idx * 16, O + (size_t)(row0 + r) * DMODEL + kk + c8 * 8);
    }
#pragma unroll
    for (int t = 0; t < 16; ++t) {
      int idx = tid + t * 256;
      int c = idx >> 5, r = idx & 31;
      BsT[idx] = (_Float16)w[(size_t)(kk + r) * DMODEL + col0 + c];
    }
    wait_async0();
    __syncthreads();
#pragma unroll
    for (int mi = 0; mi < 2; ++mi) {
      v16h a = load_a_frag(As + (wm * 32 + mi * 16) * 32, 32, lane);
#pragma unroll
      for (int ni = 0; ni < 4; ++ni) {
        v16h b = load_b_frag(BsT + (wn * 64 + ni * 16) * 32, 32, lane);
        acc[mi][ni] = wmma_f16(a, b, acc[mi][ni]);
      }
    }
  }

  const int rowoff = (lane & 16) ? 8 : 0;
#pragma unroll
  for (int mi = 0; mi < 2; ++mi) {
#pragma unroll
    for (int ni = 0; ni < 4; ++ni) {
      int colg = col0 + wn * 64 + ni * 16 + (lane & 15);
      float bv = bias[colg];
#pragma unroll
      for (int r = 0; r < 8; ++r) {
        int rowg = row0 + wm * 32 + mi * 16 + rowoff + r;
        out[(size_t)rowg * DMODEL + colg] = acc[mi][ni][r] + bv;
      }
    }
  }
}

// ---------------------------------------------------------------------------
// Host launcher.
// ---------------------------------------------------------------------------
extern "C" void kernel_launch(void* const* d_in, const int* in_sizes, int n_in,
                              void* d_out, int out_size, void* d_ws, size_t ws_size,
                              hipStream_t stream) {
  const float* x      = (const float*)d_in[0];
  const float* w_qkv  = (const float*)d_in[1];
  const float* conv_w = (const float*)d_in[2];
  const float* conv_b = (const float*)d_in[3];
  const float* gamma  = (const float*)d_in[4];
  const float* beta   = (const float*)d_in[5];
  const float* w_out  = (const float*)d_in[6];
  const float* b_out  = (const float*)d_in[7];
  float* out = (float*)d_out;

  char* wsp = (char*)d_ws;
  size_t off = 0;
  auto take = [&](size_t bytes) -> char* {
    char* p = wsp + off;
    off += (bytes + 255) & ~(size_t)255;
    return p;
  };
  const size_t QKV_ELEMS = (size_t)NB * NH * NS * DH;
  const size_t S_ELEMS   = (size_t)NB * NH * NS * NS;
  const size_t ROWS      = (size_t)NB * NH * NS;
  const size_t O_ELEMS   = (size_t)NB * NS * DMODEL;

  _Float16* qg    = (_Float16*)take(QKV_ELEMS * 2);
  _Float16* kg    = (_Float16*)take(QKV_ELEMS * 2);
  _Float16* vT    = (_Float16*)take(QKV_ELEMS * 2);
  _Float16* Sg    = (_Float16*)take(S_ELEMS * 2);
  float*    rmax  = (float*)take(ROWS * 4);
  float*    rsum  = (float*)take(ROWS * 4);
  float*    stats = (float*)take(32 * 4);
  _Float16* Og    = (_Float16*)take(O_ELEMS * 2);

  k_zero_stats<<<1, 32, 0, stream>>>(stats);
  k_qkv<<<dim3(64, 12), 256, 0, stream>>>(x, w_qkv, qg, kg, vT);
  k_scores<<<dim3(64, 16), 256, 0, stream>>>(qg, kg, Sg);
  k_rowstats<<<8192, 256, 0, stream>>>(Sg, rmax, rsum);
  k_mix<<<32768, 256, 0, stream>>>(Sg, rmax, rsum, conv_w, conv_b, stats);
  k_bnfin<<<1, 32, 0, stream>>>(stats, gamma, beta);
  k_av<<<dim3(64, 8), 256, 0, stream>>>(Sg, vT, stats, Og);
  k_proj<<<dim3(64, 4), 256, 0, stream>>>(Og, w_out, b_out, out);
}
